// SA_Layer_63436666962112
// MI455X (gfx1250) — matrix operand order repaired and checked by reading, not compiled, not run on hardware
//
#include <hip/hip_runtime.h>
#include <hip/hip_bf16.h>

#define B_ 16
#define C_ 256
#define N_ 2048

typedef __attribute__((ext_vector_type(16))) __bf16 v16bf;
typedef __attribute__((ext_vector_type(8)))  float  v8f;

union ABTile { v16bf v; unsigned u[8]; uint4 q[2]; unsigned short s[16]; };

__device__ __forceinline__ unsigned short f2bf(float f) {
  unsigned u = __float_as_uint(f);
  u += 0x7FFFu + ((u >> 16) & 1u);           // round-to-nearest-even
  return (unsigned short)(u >> 16);
}
__device__ __forceinline__ float bf2f(unsigned short h) {
  return __uint_as_float(((unsigned)h) << 16);
}

// A-matrix 16x32 bf16: lanes 0-15 row M=lane, K=0..15;
// lanes 16-31 row M=lane-16, K=16..31. Source row-major [row][K].
__device__ __forceinline__ v16bf load_a16(const unsigned short* base, int ld,
                                          int row0, int k0, int lane) {
  const int h = lane >> 4, mr = lane & 15;
  const unsigned short* p = base + (size_t)(row0 + mr) * ld + k0 + 16 * h;
  ABTile t;
  const unsigned* p0 = (const unsigned*)p;
  const unsigned* p1 = (const unsigned*)(p + 8);
  t.u[0] = p0[0]; t.u[1] = p0[1]; t.u[2] = p0[2]; t.u[3] = p0[3];
  t.u[4] = p1[0]; t.u[5] = p1[1]; t.u[6] = p1[2]; t.u[7] = p1[3];
  return t.v;
}

// B-matrix 32x16 bf16: lanes 0-15 col N=lane, K=0..15; lanes 16-31 K=16..31.
// Source is K-contiguous per column: baseT[col][K], ld in elements.
__device__ __forceinline__ v16bf load_b16(const unsigned short* baseT, int ld,
                                          int col0, int k0, int lane) {
  const int h = lane >> 4, nc = lane & 15;
  const uint4* p = (const uint4*)(baseT + (size_t)(col0 + nc) * ld + k0 + 16 * h);
  ABTile t; t.q[0] = p[0]; t.q[1] = p[1];
  return t.v;
}

// Same B-tile fetch, but from an LDS-staged 16-row slab laid out [col][K=C_].
__device__ __forceinline__ v16bf load_b16_lds(const unsigned short* slab,
                                              int k0, int lane) {
  const int h = lane >> 4, nc = lane & 15;
  const uint4* p = (const uint4*)(slab + nc * C_ + k0 + 16 * h);
  ABTile t; t.q[0] = p[0]; t.q[1] = p[1];
  return t.v;
}

__device__ __forceinline__ v8f wmma_bf(v16bf a, v16bf b, v8f c) {
  return __builtin_amdgcn_wmma_f32_16x16x32_bf16(false, a, false, b, (short)0, c,
                                                 false, false);
}

// D tile (rows r+8*half per VGPR, col = lane&15) stored transposed bf16:
// dstT[col][row] — 8 contiguous bf16 = one 16B store per lane.
__device__ __forceinline__ void store_d_tr(unsigned short* baseT, int ld,
                                           int row0, int col0, int lane, v8f d) {
  const int h = lane >> 4, c = lane & 15;
  union { uint4 q; unsigned short s[8]; } o;
#pragma unroll
  for (int r = 0; r < 8; ++r) o.s[r] = f2bf(d[r]);
  *(uint4*)(baseT + (size_t)(col0 + c) * ld + row0 + 8 * h) = o.q;
}

// ---------------- kernel 0: fp32 -> bf16 conversions ----------------
__global__ void k_convert(const float* __restrict__ x, const float* __restrict__ Wq,
                          const float* __restrict__ Wv, const float* __restrict__ Wt,
                          unsigned short* __restrict__ xt,   // [B][N][C]
                          unsigned short* __restrict__ wqb,
                          unsigned short* __restrict__ wvb,
                          unsigned short* __restrict__ wtb) {
  const size_t stride = (size_t)gridDim.x * blockDim.x;
  const size_t tid = (size_t)blockIdx.x * blockDim.x + threadIdx.x;
  const size_t totx = (size_t)B_ * C_ * N_;
  for (size_t i = tid; i < totx; i += stride) {
    int n = (int)(i % N_);
    int c = (int)((i / N_) % C_);
    int b = (int)(i / ((size_t)N_ * C_));
    xt[((size_t)b * N_ + n) * C_ + c] = f2bf(x[i]);
  }
  for (size_t i = tid; i < (size_t)C_ * C_; i += stride) {
    wqb[i] = f2bf(Wq[i]);
    wvb[i] = f2bf(Wv[i]);
    wtb[i] = f2bf(Wt[i]);
  }
}

// ------- kernel 1: Q = Wq@x (store Qt[N][C]), V = Wv@x + bv ([C][N]) ------------
__global__ void k_proj(const unsigned short* __restrict__ xt,
                       const unsigned short* __restrict__ wqb,
                       const unsigned short* __restrict__ wvb,
                       const float* __restrict__ bv,
                       unsigned short* __restrict__ Qt,  // [B][N][C]
                       unsigned short* __restrict__ Vb)  // [B][C][N]
{
  const int lane = threadIdx.x & 31;
  const int gw = blockIdx.x * (blockDim.x >> 5) + (threadIdx.x >> 5);
  const int tilesN = N_ / 32, tilesC = C_ / 16;
  const int nt = gw % tilesN, ct = (gw / tilesN) % tilesC, b = gw / (tilesN * tilesC);
  if (b >= B_) return;
  const int o0 = ct * 16, n0 = nt * 32;
  const unsigned short* xb = xt + (size_t)b * N_ * C_;
  v8f q0 = {}, q1 = {}, v0 = {}, v1 = {};
#pragma unroll
  for (int k0 = 0; k0 < C_; k0 += 32) {
    v16bf aq = load_a16(wqb, C_, o0, k0, lane);
    v16bf av = load_a16(wvb, C_, o0, k0, lane);
    v16bf b0 = load_b16(xb, C_, n0, k0, lane);
    v16bf b1 = load_b16(xb, C_, n0 + 16, k0, lane);
    q0 = wmma_bf(aq, b0, q0);
    q1 = wmma_bf(aq, b1, q1);
    v0 = wmma_bf(av, b0, v0);
    v1 = wmma_bf(av, b1, v1);
  }
  unsigned short* qt = Qt + (size_t)b * N_ * C_;
  store_d_tr(qt, C_, o0, n0, lane, q0);
  store_d_tr(qt, C_, o0, n0 + 16, lane, q1);
  const int h = lane >> 4, nc = lane & 15;
  unsigned short* vp = Vb + (size_t)b * C_ * N_;
#pragma unroll
  for (int r = 0; r < 8; ++r) {
    int o = o0 + r + 8 * h;
    float bias = bv[o];
    vp[(size_t)o * N_ + n0 + nc]      = f2bf(v0[r] + bias);
    vp[(size_t)o * N_ + n0 + 16 + nc] = f2bf(v1[r] + bias);
  }
}

// ------- kernel 2: energy = Qt Qt^T, online row-softmax, store S^T --------------
// 32 query rows per wave (two resident A slabs). The key B-slab (16 m-rows x 256 K
// of Qt = one contiguous 8KB block) is identical for all 8 waves of the block, so
// it is staged in LDS once, double-buffered, with a global->VGPR prefetch of
// tile m+1 overlapping the 16 WMMAs of tile m.
__global__ void k_attn(const unsigned short* __restrict__ Qt,
                       unsigned short* __restrict__ St)  // [B][m][n] (S transposed)
{
  __shared__ unsigned short sbuf[2][16 * C_];            // 2 x 8KB
  const int tid = threadIdx.x;
  const int lane = tid & 31;
  const int gw = blockIdx.x * (blockDim.x >> 5) + (tid >> 5);
  const int nt = gw % (N_ / 32), b = gw / (N_ / 32);     // all waves: same b
  const int n0 = nt * 32;
  const unsigned short* q = Qt + (size_t)b * N_ * C_;

  v16bf a[2][8];  // resident A-tiles for rows [n0,n0+16) and [n0+16,n0+32)
#pragma unroll
  for (int k = 0; k < 8; ++k) {
    a[0][k] = load_a16(q, C_, n0, k * 32, lane);
    a[1][k] = load_a16(q, C_, n0 + 16, k * 32, lane);
  }

  float lmax[2][8], lsum[2][8];
#pragma unroll
  for (int g = 0; g < 2; ++g)
#pragma unroll
    for (int r = 0; r < 8; ++r) { lmax[g][r] = -3.0e38f; lsum[g][r] = 0.0f; }

  const int h = lane >> 4, mc = lane & 15;
  unsigned short* s = St + (size_t)b * N_ * N_;

  for (int pass = 0; pass < 2; ++pass) {
    // stage tile m0=0 into buffer 0 (one contiguous 8KB slab, 2 uint4/thread)
    {
      const uint4* src = (const uint4*)q;
      uint4* dst = (uint4*)sbuf[0];
      dst[tid]       = src[tid];
      dst[tid + 256] = src[tid + 256];
    }
    __syncthreads();

    for (int m0 = 0; m0 < N_; m0 += 16) {
      const int cur = (m0 >> 4) & 1;
      // prefetch next slab into registers while computing on the current one
      uint4 pre0, pre1;
      const bool more = (m0 + 16) < N_;
      if (more) {
        const uint4* src = (const uint4*)(q + (size_t)(m0 + 16) * C_);
        pre0 = src[tid];
        pre1 = src[tid + 256];
      }

      v8f acc0 = {}, acc1 = {};
#pragma unroll
      for (int k = 0; k < 8; ++k) {
        v16bf bm = load_b16_lds(sbuf[cur], k * 32, lane);
        acc0 = wmma_bf(a[0][k], bm, acc0);
        acc1 = wmma_bf(a[1][k], bm, acc1);
      }

      if (pass == 0) {
        // online max/sum over m
#pragma unroll
        for (int r = 0; r < 8; ++r) {
          float e0 = acc0[r], nm0 = fmaxf(lmax[0][r], e0);
          lsum[0][r] = lsum[0][r] * __expf(lmax[0][r] - nm0) + __expf(e0 - nm0);
          lmax[0][r] = nm0;
          float e1 = acc1[r], nm1 = fmaxf(lmax[1][r], e1);
          lsum[1][r] = lsum[1][r] * __expf(lmax[1][r] - nm1) + __expf(e1 - nm1);
          lmax[1][r] = nm1;
        }
      } else {
        // normalize and write S^T[m][n] in bf16
        union { uint4 q4; unsigned short us[8]; } o0, o1;
#pragma unroll
        for (int r = 0; r < 8; ++r) {
          o0.us[r] = f2bf(__expf(acc0[r] - lmax[0][r]) * lsum[0][r]);
          o1.us[r] = f2bf(__expf(acc1[r] - lmax[1][r]) * lsum[1][r]);
        }
        unsigned short* row = s + (size_t)(m0 + mc) * N_;
        *(uint4*)(row + n0 + 8 * h)      = o0.q4;
        *(uint4*)(row + n0 + 16 + 8 * h) = o1.q4;
      }

      if (more) {
        uint4* dst = (uint4*)sbuf[cur ^ 1];
        dst[tid]       = pre0;
        dst[tid + 256] = pre1;
      }
      __syncthreads();
    }

    if (pass == 0) {
      // combine across the 16 lanes that share each row (masks stay in halves)
#pragma unroll
      for (int mask = 1; mask < 16; mask <<= 1) {
#pragma unroll
        for (int g = 0; g < 2; ++g)
#pragma unroll
          for (int r = 0; r < 8; ++r) {
            float om = __shfl_xor(lmax[g][r], mask, 32);
            float os = __shfl_xor(lsum[g][r], mask, 32);
            float nm = fmaxf(lmax[g][r], om);
            lsum[g][r] = lsum[g][r] * __expf(lmax[g][r] - nm) + os * __expf(om - nm);
            lmax[g][r] = nm;
          }
      }
      // reuse lsum as 1/rowsum for pass 2
#pragma unroll
      for (int g = 0; g < 2; ++g)
#pragma unroll
        for (int r = 0; r < 8; ++r) lsum[g][r] = 1.0f / lsum[g][r];
    }
  }
}

// ------- kernel 3: x_r = V @ S / colsum; D = x - x_r (store Dt[N][C]) -----------
// 32(c) x 32(m) per wave: 2 A + 2 B loads feed 4 WMMAs per k-chunk.
__global__ void k_xr(const unsigned short* __restrict__ Vb,
                     const unsigned short* __restrict__ St,
                     const float* __restrict__ x,
                     unsigned short* __restrict__ Dt)  // [B][n][c]
{
  const int lane = threadIdx.x & 31;
  const int gw = blockIdx.x * (blockDim.x >> 5) + (threadIdx.x >> 5);
  const int tilesN = N_ / 32, tilesC = C_ / 32;
  const int mt = gw % tilesN, ct = (gw / tilesN) % tilesC, b = gw / (tilesN * tilesC);
  if (b >= B_) return;
  const int c0 = ct * 32, m0 = mt * 32;
  const unsigned short* v = Vb + (size_t)b * C_ * N_;
  const unsigned short* s = St + (size_t)b * N_ * N_;

  v8f acc00 = {}, acc01 = {}, acc10 = {}, acc11 = {};
  float cs0 = 0.0f, cs1 = 0.0f;  // per-lane partial colsums for the two m groups
  const int h = lane >> 4, mc = lane & 15;
  for (int k0 = 0; k0 < N_; k0 += 32) {
    v16bf a0 = load_a16(v, N_, c0, k0, lane);
    v16bf a1 = load_a16(v, N_, c0 + 16, k0, lane);
    const uint4* bp0 = (const uint4*)(s + (size_t)(m0 + mc) * N_ + k0 + 16 * h);
    const uint4* bp1 = (const uint4*)(s + (size_t)(m0 + 16 + mc) * N_ + k0 + 16 * h);
    ABTile b0, b1;
    b0.q[0] = bp0[0]; b0.q[1] = bp0[1];
    b1.q[0] = bp1[0]; b1.q[1] = bp1[1];
#pragma unroll
    for (int j = 0; j < 16; ++j) { cs0 += bf2f(b0.s[j]); cs1 += bf2f(b1.s[j]); }
    acc00 = wmma_bf(a0, b0.v, acc00);
    acc01 = wmma_bf(a0, b1.v, acc01);
    acc10 = wmma_bf(a1, b0.v, acc10);
    acc11 = wmma_bf(a1, b1.v, acc11);
  }
  // lane L and L+16 cover complementary halves of the n-range for each column
  cs0 += __shfl_xor(cs0, 16, 32);
  cs1 += __shfl_xor(cs1, 16, 32);
  const float rcp0 = 1.0f / (1e-9f + cs0);
  const float rcp1 = 1.0f / (1e-9f + cs1);

  unsigned short* dt = Dt + (size_t)b * N_ * C_;
  v8f d00, d01, d10, d11;
#pragma unroll
  for (int r = 0; r < 8; ++r) {
    int ca = c0 + r + 8 * h, cb = ca + 16;
    int ma = m0 + mc, mb = ma + 16;
    d00[r] = x[((size_t)b * C_ + ca) * N_ + ma] - acc00[r] * rcp0;
    d01[r] = x[((size_t)b * C_ + ca) * N_ + mb] - acc01[r] * rcp1;
    d10[r] = x[((size_t)b * C_ + cb) * N_ + ma] - acc10[r] * rcp0;
    d11[r] = x[((size_t)b * C_ + cb) * N_ + mb] - acc11[r] * rcp1;
  }
  store_d_tr(dt, C_, c0,      m0,      lane, d00);
  store_d_tr(dt, C_, c0,      m0 + 16, lane, d01);
  store_d_tr(dt, C_, c0 + 16, m0,      lane, d10);
  store_d_tr(dt, C_, c0 + 16, m0 + 16, lane, d11);
}

// ------- kernel 4: out = x + relu(Wt @ D + bt) ----------------------------------
// 32(o) x 16(n) per wave: two A-tiles share each B-tile load.
__global__ void k_final(const unsigned short* __restrict__ Dt,
                        const unsigned short* __restrict__ wtb,
                        const float* __restrict__ btv,
                        const float* __restrict__ x,
                        float* __restrict__ out) {
  const int lane = threadIdx.x & 31;
  const int gw = blockIdx.x * (blockDim.x >> 5) + (threadIdx.x >> 5);
  const int tilesN = N_ / 16, tilesC = C_ / 32;
  const int nt = gw % tilesN, ct = (gw / tilesN) % tilesC, b = gw / (tilesN * tilesC);
  if (b >= B_) return;
  const int o0 = ct * 32, n0 = nt * 16;
  const unsigned short* db = Dt + (size_t)b * N_ * C_;
  v8f acc0 = {}, acc1 = {};
#pragma unroll
  for (int k0 = 0; k0 < C_; k0 += 32) {
    v16bf bd = load_b16(db, C_, n0, k0, lane);
    v16bf a0 = load_a16(wtb, C_, o0, k0, lane);
    v16bf a1 = load_a16(wtb, C_, o0 + 16, k0, lane);
    acc0 = wmma_bf(a0, bd, acc0);
    acc1 = wmma_bf(a1, bd, acc1);
  }
  const int h = lane >> 4, nc = lane & 15;
#pragma unroll
  for (int r = 0; r < 8; ++r) {
    int oa = o0 + r + 8 * h, ob = oa + 16, n = n0 + nc;
    size_t ia = ((size_t)b * C_ + oa) * N_ + n;
    size_t ib = ((size_t)b * C_ + ob) * N_ + n;
    out[ia] = x[ia] + fmaxf(acc0[r] + btv[oa], 0.0f);
    out[ib] = x[ib] + fmaxf(acc1[r] + btv[ob], 0.0f);
  }
}

// ---------------- launcher ------------------------------------------------------
extern "C" void kernel_launch(void* const* d_in, const int* in_sizes, int n_in,
                              void* d_out, int out_size, void* d_ws, size_t ws_size,
                              hipStream_t stream) {
  const float* x  = (const float*)d_in[0];
  const float* Wq = (const float*)d_in[1];
  const float* Wv = (const float*)d_in[2];
  const float* bv = (const float*)d_in[3];
  const float* Wt = (const float*)d_in[4];
  const float* bt = (const float*)d_in[5];
  float* out = (float*)d_out;

  char* ws = (char*)d_ws;
  size_t off = 0;
  unsigned short* xt  = (unsigned short*)(ws + off); off += (size_t)B_ * N_ * C_ * 2;
  unsigned short* wqb = (unsigned short*)(ws + off); off += (size_t)C_ * C_ * 2;
  unsigned short* wvb = (unsigned short*)(ws + off); off += (size_t)C_ * C_ * 2;
  unsigned short* wtb = (unsigned short*)(ws + off); off += (size_t)C_ * C_ * 2;
  unsigned short* Qt  = (unsigned short*)(ws + off); off += (size_t)B_ * N_ * C_ * 2;
  unsigned short* Vb  = (unsigned short*)(ws + off); off += (size_t)B_ * C_ * N_ * 2;
  unsigned short* St  = (unsigned short*)(ws + off); off += (size_t)B_ * N_ * N_ * 2;
  unsigned short* Dt  = (unsigned short*)(ws + off); off += (size_t)B_ * N_ * C_ * 2;
  (void)ws_size; (void)in_sizes; (void)n_in; (void)out_size;

  const int THREADS = 256;               // 8 wave32 per block
  const int WPB = THREADS / 32;

  k_convert<<<2048, THREADS, 0, stream>>>(x, Wq, Wv, Wt, xt, wqb, wvb, wtb);

  int wproj = B_ * (C_ / 16) * (N_ / 32);               // 16384 waves
  k_proj<<<wproj / WPB, THREADS, 0, stream>>>(xt, wqb, wvb, bv, Qt, Vb);

  int wattn = B_ * (N_ / 32);                           // 1024 waves, 128 blocks
  k_attn<<<wattn / WPB, THREADS, 0, stream>>>(Qt, St);

  int wxr = B_ * (C_ / 32) * (N_ / 32);                 // 8192 waves
  k_xr<<<wxr / WPB, THREADS, 0, stream>>>(Vb, St, x, Dt);

  int wfin = B_ * (C_ / 32) * (N_ / 16);                // 16384 waves
  k_final<<<wfin / WPB, THREADS, 0, stream>>>(Dt, wtb, bt, x, out);
}
